// _EmbeddingActionScoringNetwork_41953240547573
// MI455X (gfx1250) — matrix operand order repaired and checked by reading, not compiled, not run on hardware
//
#include <hip/hip_runtime.h>
#include <stdint.h>

// ---------------------------------------------------------------------------
// Problem constants (from reference)
#define BB   2048
#define NN   64
#define DD   64
#define HH   256
#define MACT (BB * NN)          // 131072 action rows
#define KS   672                // 662 state-in, padded to x32
#define KA   480                // 464 action-in, padded to x32
#define KC   512                // combined K (already x32)

// ---------------------------------------------------------------------------
// CDNA5 WMMA types / fragment helpers (wave32, 16x16x32 bf16 -> f32)
typedef __attribute__((ext_vector_type(16))) __bf16 v16bf;
typedef __attribute__((ext_vector_type(8)))  float  v8f;

struct __attribute__((aligned(16))) U128 { unsigned int x, y, z, w; };
union FragU { U128 u[2]; v16bf v; };

// A tile in LDS: [rows][40 bf16 stride], row-major.
// ISA layout: lane holds row M=lane&15; VGPR pairs hold K = (e>>3)*16 + ((lane>>4)<<3) + (e&7)
__device__ __forceinline__ v16bf ld_frag_a(const __bf16* As_rows, int lane) {
  int row = lane & 15;
  int kb  = (lane >> 4) << 3;                  // 0 or 8
  const __bf16* p = As_rows + row * 40 + kb;
  FragU f;
  f.u[0] = *(const U128*)p;                    // K = kb .. kb+7
  f.u[1] = *(const U128*)(p + 16);             // K = 16+kb .. 16+kb+7
  return f.v;
}

// B staged transposed in LDS: BsT[n][k], stride 40.
// ISA layout: lane holds col N=lane&15; elements K = (e>>3)*16 + ((lane>>4)<<3) + (e&7)
__device__ __forceinline__ v16bf ld_frag_b(const __bf16* BsT, int lane, int nt) {
  int n  = (nt << 4) + (lane & 15);
  int kb = (lane >> 4) << 3;                   // 0 or 8
  const __bf16* p = BsT + n * 40 + kb;
  FragU f;
  f.u[0] = *(const U128*)p;
  f.u[1] = *(const U128*)(p + 16);
  return f.v;
}

// Stage one 32xK-chunk of a bf16 weight matrix W[K][256] into BsT[256][40]
__device__ __forceinline__ void stage_b(__bf16* BsT, const __bf16* W, int kc, int tid) {
  const __bf16* src = W + (((size_t)kc) << 5) * 256 + tid;   // col n = tid
  #pragma unroll 8
  for (int k = 0; k < 32; ++k) BsT[tid * 40 + k] = src[(size_t)k * 256];
}

#define WMMA_BF16(acc, a, b) \
  __builtin_amdgcn_wmma_f32_16x16x32_bf16(false, (a), false, (b), (short)0, (acc), false, false)

// One K-chunk for a 32x128 wave tile: 2 A-frags + 8 B-frags feed 16 WMMAs,
// so every LDS fragment load is amortized over >=2 matrix ops.  B fragments
// are software-pipelined one tile ahead.
__device__ __forceinline__ void wmma_sweep(v8f acc[][8], const __bf16* As,
                                           const __bf16* BsT, int mwave, int nwave, int lane) {
  v16bf a0 = ld_frag_a(As + (mwave * 32) * 40, lane);
  v16bf a1 = ld_frag_a(As + (mwave * 32 + 16) * 40, lane);
  v16bf b = ld_frag_b(BsT, lane, nwave * 8);
  #pragma unroll
  for (int nt = 0; nt < 8; ++nt) {
    v16bf bn = b;
    if (nt < 7) bn = ld_frag_b(BsT, lane, nwave * 8 + nt + 1);
    acc[0][nt] = WMMA_BF16(acc[0][nt], a0, b);
    acc[1][nt] = WMMA_BF16(acc[1][nt], a1, b);
    b = bn;
  }
}

// ---------------------------------------------------------------------------
// Prep kernel: f32 weight [Ksrc][256] -> bf16 [Kdst][256] with zero K-padding.
__global__ __launch_bounds__(256) void cvt_w(const float* __restrict__ src,
                                             __bf16* __restrict__ dst, int Ksrc) {
  int k = blockIdx.x, n = threadIdx.x;
  size_t idx = (size_t)k * 256 + n;
  dst[idx] = (k < Ksrc) ? (__bf16)src[idx] : (__bf16)0.f;
}

// ---------------------------------------------------------------------------
// State feature encoder: one wave per batch row -> Sin[b][672] bf16 (zero pad).
__device__ __forceinline__ void pool_emb(__bf16* dst, const float* emb,
                                         const int* ids, const int* mask, int L, int lane) {
  float cnt = 0.f;
  for (int l = 0; l < L; ++l) cnt += (float)mask[l];
  float inv = 1.f / fmaxf(cnt, 1.f);
  for (int d = lane; d < 64; d += 32) {
    float s = 0.f;
    for (int l = 0; l < L; ++l)
      if (mask[l]) s += emb[ids[l] * 64 + d];
    dst[d] = (__bf16)(s * inv);
  }
}

__device__ __forceinline__ void pool_emb_sc(__bf16* dst, const float* emb,
                                            const float* sc, int SC,
                                            const int* ids, const int* mask, int L, int lane) {
  float cnt = 0.f;
  for (int l = 0; l < L; ++l) cnt += (float)mask[l];
  float inv = 1.f / fmaxf(cnt, 1.f);
  for (int d = lane; d < 64 + SC; d += 32) {
    float s = 0.f;
    for (int l = 0; l < L; ++l)
      if (mask[l]) s += (d < 64) ? emb[ids[l] * 64 + d] : sc[l * SC + (d - 64)];
    dst[d] = (__bf16)(s * inv);
  }
}

__global__ __launch_bounds__(256) void encode_state(
    const float* __restrict__ state_scalars,
    const int* __restrict__ mode_id, const int* __restrict__ terrain_id,
    const int* __restrict__ site_type_id,
    const int* __restrict__ hand_ids, const int* __restrict__ hand_mask,
    const int* __restrict__ unit_ids, const int* __restrict__ unit_mask,
    const int* __restrict__ ce_ids, const float* __restrict__ ce_sc, const int* __restrict__ ce_mask,
    const int* __restrict__ skill_ids, const int* __restrict__ skill_mask,
    const int* __restrict__ vs_ids, const float* __restrict__ vs_sc, const int* __restrict__ vs_mask,
    const int* __restrict__ me_ids, const float* __restrict__ me_sc, const int* __restrict__ me_mask,
    const float* __restrict__ card_emb, const float* __restrict__ unit_emb,
    const float* __restrict__ enemy_emb, const float* __restrict__ mode_emb,
    const float* __restrict__ terr_emb, const float* __restrict__ site_emb,
    const float* __restrict__ skill_emb, const float* __restrict__ msite_emb,
    __bf16* __restrict__ Sin) {
  int b = blockIdx.x * 8 + (threadIdx.x >> 5);
  int lane = threadIdx.x & 31;
  __bf16* out = Sin + (size_t)b * KS;
  for (int d = lane; d < 64; d += 32) {
    out[0   + d] = (__bf16)state_scalars[b * 64 + d];
    out[64  + d] = (__bf16)mode_emb[mode_id[b] * 64 + d];
    out[256 + d] = (__bf16)terr_emb[terrain_id[b] * 64 + d];
    out[320 + d] = (__bf16)site_emb[site_type_id[b] * 64 + d];
  }
  pool_emb(out + 128, card_emb,  hand_ids + b * 16, hand_mask + b * 16, 16, lane);
  pool_emb(out + 192, unit_emb,  unit_ids + b * 8,  unit_mask + b * 8,  8,  lane);
  pool_emb_sc(out + 384, enemy_emb, ce_sc + (size_t)b * 64, 8, ce_ids + b * 8,  ce_mask + b * 8,  8,  lane);
  pool_emb(out + 456, skill_emb, skill_ids + b * 8, skill_mask + b * 8, 8,  lane);
  pool_emb_sc(out + 520, msite_emb, vs_sc + (size_t)b * 96, 6, vs_ids + b * 16, vs_mask + b * 16, 16, lane);
  pool_emb_sc(out + 590, enemy_emb, me_sc + (size_t)b * 128, 8, me_ids + b * 16, me_mask + b * 16, 16, lane);
  for (int d = 662 + lane; d < KS; d += 32) out[d] = (__bf16)0.f;
}

// ---------------------------------------------------------------------------
// Target pool: tpool[m][d] = masked mean over 8 target enemies (bf16).
__global__ __launch_bounds__(256) void pool_targets(const float* __restrict__ enemy_emb,
                                                    const int* __restrict__ tids,
                                                    const int* __restrict__ tmask,
                                                    __bf16* __restrict__ tp) {
  int idx = blockIdx.x * 256 + threadIdx.x;
  int m = idx >> 6, d = idx & 63;
  const int* id = tids + (size_t)m * 8;
  const int* mk = tmask + (size_t)m * 8;
  float cnt = 0.f, s = 0.f;
  #pragma unroll
  for (int t = 0; t < 8; ++t)
    if (mk[t]) { cnt += 1.f; s += enemy_emb[id[t] * 64 + d]; }
  tp[idx] = (__bf16)(s / fmaxf(cnt, 1.f));
}

// ---------------------------------------------------------------------------
// GEMM 1: state_repr = tanh(Sin @ Ws + bs).  M=2048, K=672, N=256.
// Double-buffered LDS staging; 8 waves tiled 4(M) x 2(N), 32x128 per wave.
__global__ __launch_bounds__(256) void gemm_state(const __bf16* __restrict__ Sin,
                                                  const __bf16* __restrict__ Wbf,
                                                  const float* __restrict__ bias,
                                                  float* __restrict__ outf,
                                                  __bf16* __restrict__ outbf) {
  __shared__ __attribute__((aligned(16))) __bf16 As[2][128 * 40];
  __shared__ __attribute__((aligned(16))) __bf16 BsT[2][256 * 40];
  int tid = threadIdx.x, lane = tid & 31, w = tid >> 5;
  int mwave = w >> 1, nwave = w & 1;
  int m0 = blockIdx.x * 128;
  int r = tid >> 1, ch = (tid & 1) << 4;
  v8f acc[2][8] = {};

  auto stageA = [&](int kc, int buf) {
    const U128* s = (const U128*)(Sin + (size_t)(m0 + r) * KS + (kc << 5) + ch);
    U128* d = (U128*)(&As[buf][r * 40 + ch]);
    d[0] = s[0]; d[1] = s[1];
  };

  stageA(0, 0);
  stage_b(BsT[0], Wbf, 0, tid);
  __syncthreads();
  #pragma unroll 2
  for (int kc = 0; kc < KS / 32; ++kc) {
    int cur = kc & 1;
    if (kc + 1 < KS / 32) {                 // prefetch next chunk into other buffer
      stageA(kc + 1, cur ^ 1);
      stage_b(BsT[cur ^ 1], Wbf, kc + 1, tid);
    }
    wmma_sweep(acc, As[cur], BsT[cur], mwave, nwave, lane);
    __syncthreads();
  }

  int nlo = lane & 15, hi = lane >> 4;
  #pragma unroll
  for (int i = 0; i < 2; ++i)
  #pragma unroll
  for (int nt = 0; nt < 8; ++nt) {
    int n = nwave * 128 + (nt << 4) + nlo;
    float bv = bias[n];
    #pragma unroll
    for (int rr = 0; rr < 8; ++rr) {
      int m = m0 + mwave * 32 + i * 16 + hi * 8 + rr;
      float v = tanhf(acc[i][nt][rr] + bv);
      outf[(size_t)m * 256 + n]  = v;
      outbf[(size_t)m * 256 + n] = (__bf16)v;
    }
  }
}

// ---------------------------------------------------------------------------
// GEMM 2: action_repr = tanh(action_input @ Wa + ba); action_input rows built
// in LDS from embedding gathers.  M=131072, K=480, N=256.
__global__ __launch_bounds__(256) void gemm_action(
    const int* __restrict__ action_ids, const float* __restrict__ action_sc,
    const __bf16* __restrict__ tpool,
    const float* __restrict__ at_emb, const float* __restrict__ src_emb,
    const float* __restrict__ card_emb, const float* __restrict__ unit_emb,
    const float* __restrict__ enemy_emb, const float* __restrict__ skill_emb,
    const __bf16* __restrict__ Wbf, const float* __restrict__ bias,
    __bf16* __restrict__ arepr) {
  __shared__ __attribute__((aligned(16))) __bf16 As[2][128 * 40];
  __shared__ __attribute__((aligned(16))) __bf16 BsT[2][256 * 40];
  int tid = threadIdx.x, lane = tid & 31, w = tid >> 5;
  int mwave = w >> 1, nwave = w & 1;
  int m0 = blockIdx.x * 128;
  int r = tid >> 1, ch = (tid & 1) << 4;   // each thread fills 16 contiguous K of one row
  int m = m0 + r;
  v8f acc[2][8] = {};

  auto stageA = [&](int kc, int buf) {
    __bf16* dst = &As[buf][r * 40 + ch];
    if (kc < 12) {                         // 6 embedding segments, 64-wide each
      int seg = kc >> 1;
      const float* tab = seg == 0 ? at_emb   : seg == 1 ? src_emb :
                         seg == 2 ? card_emb : seg == 3 ? unit_emb :
                         seg == 4 ? enemy_emb : skill_emb;
      int idx = action_ids[(size_t)m * 6 + seg];
      const float* e = tab + idx * 64 + ((kc & 1) << 5) + ch;
      #pragma unroll
      for (int j = 0; j < 16; ++j) dst[j] = (__bf16)e[j];
    } else if (kc < 14) {                  // target pool segment (bf16, contiguous)
      const U128* s = (const U128*)(tpool + (size_t)m * 64 + ((kc - 12) << 5) + ch);
      U128* d = (U128*)dst;
      d[0] = s[0]; d[1] = s[1];
    } else {                               // scalars (16) + zero pad (16)
      if (ch == 0) {
        const float* s = action_sc + (size_t)m * 16;
        #pragma unroll
        for (int j = 0; j < 16; ++j) dst[j] = (__bf16)s[j];
      } else {
        #pragma unroll
        for (int j = 0; j < 16; ++j) dst[j] = (__bf16)0.f;
      }
    }
  };

  stageA(0, 0);
  stage_b(BsT[0], Wbf, 0, tid);
  __syncthreads();
  #pragma unroll 2
  for (int kc = 0; kc < KA / 32; ++kc) {
    int cur = kc & 1;
    if (kc + 1 < KA / 32) {
      stageA(kc + 1, cur ^ 1);
      stage_b(BsT[cur ^ 1], Wbf, kc + 1, tid);
    }
    wmma_sweep(acc, As[cur], BsT[cur], mwave, nwave, lane);
    __syncthreads();
  }

  int nlo = lane & 15, hi = lane >> 4;
  #pragma unroll
  for (int i = 0; i < 2; ++i)
  #pragma unroll
  for (int nt = 0; nt < 8; ++nt) {
    int n = nwave * 128 + (nt << 4) + nlo;
    float bv = bias[n];
    #pragma unroll
    for (int rr = 0; rr < 8; ++rr) {
      int mm = m0 + mwave * 32 + i * 16 + hi * 8 + rr;
      arepr[(size_t)mm * 256 + n] = (__bf16)tanhf(acc[i][nt][rr] + bv);
    }
  }
}

// ---------------------------------------------------------------------------
// GEMM 3: logits = tanh([state_b, action_repr] @ W1 + b1) @ W2 + b2.
// M=131072, K=512, N=256; W2 reduction fused: intra-wave 16-lane butterflies,
// then a tiny LDS combine across the two N-halves (waves nwave 0/1).
__global__ __launch_bounds__(256) void gemm_comb(
    const __bf16* __restrict__ srepr_bf, const __bf16* __restrict__ arepr,
    const __bf16* __restrict__ Wbf, const float* __restrict__ b1,
    const float* __restrict__ W2, const float* __restrict__ b2,
    float* __restrict__ logits) {
  __shared__ __attribute__((aligned(16))) __bf16 As[2][128 * 40];
  __shared__ __attribute__((aligned(16))) __bf16 BsT[2][256 * 40];
  int tid = threadIdx.x, lane = tid & 31, w = tid >> 5;
  int mwave = w >> 1, nwave = w & 1;
  int m0 = blockIdx.x * 128;
  int r = tid >> 1, ch = (tid & 1) << 4;
  int m = m0 + r;
  v8f acc[2][8] = {};

  auto stageA = [&](int kc, int buf) {
    const __bf16* s;
    if (kc < 8) s = srepr_bf + (((size_t)(m >> 6)) << 8) + (kc << 5) + ch;   // state half
    else        s = arepr    + (((size_t)m) << 8) + ((kc - 8) << 5) + ch;    // action half
    U128* d = (U128*)(&As[buf][r * 40 + ch]);
    const U128* su = (const U128*)s;
    d[0] = su[0]; d[1] = su[1];
  };

  stageA(0, 0);
  stage_b(BsT[0], Wbf, 0, tid);
  __syncthreads();
  #pragma unroll 2
  for (int kc = 0; kc < KC / 32; ++kc) {
    int cur = kc & 1;
    if (kc + 1 < KC / 32) {
      stageA(kc + 1, cur ^ 1);
      stage_b(BsT[cur ^ 1], Wbf, kc + 1, tid);
    }
    wmma_sweep(acc, As[cur], BsT[cur], mwave, nwave, lane);
    __syncthreads();
  }

  // Fused epilogue: h = tanh(acc + b1); partial = h * W2[n].
  int nlo = lane & 15, hi = lane >> 4;
  float p[2][8] = {};
  #pragma unroll
  for (int i = 0; i < 2; ++i)
  #pragma unroll
  for (int nt = 0; nt < 8; ++nt) {
    int n = nwave * 128 + (nt << 4) + nlo;
    float bv = b1[n], w2 = W2[n];
    #pragma unroll
    for (int rr = 0; rr < 8; ++rr) p[i][rr] += tanhf(acc[i][nt][rr] + bv) * w2;
  }
  // 16-lane butterfly per row, then cross-wave (N-half) combine via LDS.
  float* part = (float*)&As[0][0];   // reuse LDS after final barrier: part[2][128]
  #pragma unroll
  for (int i = 0; i < 2; ++i)
  #pragma unroll
  for (int rr = 0; rr < 8; ++rr) {
    float s = p[i][rr];
    s += __shfl_xor(s, 1, 32);
    s += __shfl_xor(s, 2, 32);
    s += __shfl_xor(s, 4, 32);
    s += __shfl_xor(s, 8, 32);
    if (nlo == 0) part[nwave * 128 + mwave * 32 + i * 16 + hi * 8 + rr] = s;
  }
  __syncthreads();
  if (tid < 128) logits[m0 + tid] = part[tid] + part[128 + tid] + b2[0];
}

// ---------------------------------------------------------------------------
// Value head: value[b] = state_repr_f32[b] . W_v + b_v  (one wave per row)
__global__ __launch_bounds__(256) void value_head(const float* __restrict__ srepr,
                                                  const float* __restrict__ Wv,
                                                  const float* __restrict__ bv,
                                                  float* __restrict__ out) {
  int b = blockIdx.x * 8 + (threadIdx.x >> 5);
  int lane = threadIdx.x & 31;
  float s = 0.f;
  for (int h = lane; h < 256; h += 32) s += srepr[(size_t)b * 256 + h] * Wv[h];
  #pragma unroll
  for (int off = 16; off; off >>= 1) s += __shfl_xor(s, off, 32);
  if (lane == 0) out[b] = s + bv[0];
}

// ---------------------------------------------------------------------------
extern "C" void kernel_launch(void* const* d_in, const int* in_sizes, int n_in,
                              void* d_out, int out_size, void* d_ws, size_t ws_size,
                              hipStream_t stream) {
  const float* state_scalars = (const float*)d_in[0];
  const int*   mode_id       = (const int*)d_in[1];
  const int*   terrain_id    = (const int*)d_in[2];
  const int*   site_type_id  = (const int*)d_in[3];
  const int*   hand_ids      = (const int*)d_in[4];
  const int*   hand_mask     = (const int*)d_in[5];
  const int*   unit_ids      = (const int*)d_in[6];
  const int*   unit_mask     = (const int*)d_in[7];
  const int*   ce_ids        = (const int*)d_in[8];
  const float* ce_sc         = (const float*)d_in[9];
  const int*   ce_mask       = (const int*)d_in[10];
  const int*   skill_ids     = (const int*)d_in[11];
  const int*   skill_mask    = (const int*)d_in[12];
  const int*   vs_ids        = (const int*)d_in[13];
  const float* vs_sc         = (const float*)d_in[14];
  const int*   vs_mask       = (const int*)d_in[15];
  const int*   me_ids        = (const int*)d_in[16];
  const float* me_sc         = (const float*)d_in[17];
  const int*   me_mask       = (const int*)d_in[18];
  const int*   action_ids    = (const int*)d_in[19];
  const float* action_sc     = (const float*)d_in[20];
  const int*   tgt_ids       = (const int*)d_in[21];
  const int*   tgt_mask      = (const int*)d_in[22];
  const float* card_emb      = (const float*)d_in[23];
  const float* unit_emb      = (const float*)d_in[24];
  const float* enemy_emb     = (const float*)d_in[25];
  const float* at_emb        = (const float*)d_in[26];
  const float* src_emb       = (const float*)d_in[27];
  const float* mode_emb      = (const float*)d_in[28];
  const float* terr_emb      = (const float*)d_in[29];
  const float* site_emb      = (const float*)d_in[30];
  const float* skill_emb     = (const float*)d_in[31];
  const float* msite_emb     = (const float*)d_in[32];
  const float* W_s = (const float*)d_in[33];
  const float* b_s = (const float*)d_in[34];
  const float* W_a = (const float*)d_in[35];
  const float* b_a = (const float*)d_in[36];
  const float* W1  = (const float*)d_in[37];
  const float* b1  = (const float*)d_in[38];
  const float* W2  = (const float*)d_in[39];
  const float* b2  = (const float*)d_in[40];
  const float* W_v = (const float*)d_in[41];
  const float* b_v = (const float*)d_in[42];

  // Workspace layout (bytes, all 256B-aligned); total ~90.7 MB.
  char* ws = (char*)d_ws;
  __bf16* Ws_bf    = (__bf16*)(ws + 0);          // 672*256*2   = 344064
  __bf16* Wa_bf    = (__bf16*)(ws + 344064);     // 480*256*2   = 245760
  __bf16* W1_bf    = (__bf16*)(ws + 589824);     // 512*256*2   = 262144
  __bf16* Sin      = (__bf16*)(ws + 851968);     // 2048*672*2  = 2752512
  float*  srepr_f  = (float*) (ws + 3604480);    // 2048*256*4  = 2097152
  __bf16* srepr_bf = (__bf16*)(ws + 5701632);    // 2048*256*2  = 1048576
  __bf16* tpool    = (__bf16*)(ws + 6750208);    // 131072*64*2 = 16777216
  __bf16* arepr    = (__bf16*)(ws + 23527424);   // 131072*256*2= 67108864 -> 90636288 total

  float* logits = (float*)d_out;       // [B*N]
  float* value  = logits + MACT;       // [B]

  // Weight conversion (padded K)
  cvt_w<<<KS, 256, 0, stream>>>(W_s, Ws_bf, 662);
  cvt_w<<<KA, 256, 0, stream>>>(W_a, Wa_bf, 464);
  cvt_w<<<KC, 256, 0, stream>>>(W1,  W1_bf, 512);

  // Feature builders
  encode_state<<<BB / 8, 256, 0, stream>>>(state_scalars, mode_id, terrain_id, site_type_id,
      hand_ids, hand_mask, unit_ids, unit_mask, ce_ids, ce_sc, ce_mask,
      skill_ids, skill_mask, vs_ids, vs_sc, vs_mask, me_ids, me_sc, me_mask,
      card_emb, unit_emb, enemy_emb, mode_emb, terr_emb, site_emb, skill_emb, msite_emb, Sin);
  pool_targets<<<(MACT * 64) / 256, 256, 0, stream>>>(enemy_emb, tgt_ids, tgt_mask, tpool);

  // WMMA GEMM pipeline
  gemm_state<<<BB / 128, 256, 0, stream>>>(Sin, Ws_bf, b_s, srepr_f, srepr_bf);
  value_head<<<BB / 8, 256, 0, stream>>>(srepr_f, W_v, b_v, value);
  gemm_action<<<MACT / 128, 256, 0, stream>>>(action_ids, action_sc, tpool,
      at_emb, src_emb, card_emb, unit_emb, enemy_emb, skill_emb, Wa_bf, b_a, arepr);
  gemm_comb<<<MACT / 128, 256, 0, stream>>>(srepr_bf, arepr, W1_bf, b1, W2, b2, logits);
}
